// CapInfoNCE_45973329936914
// MI455X (gfx1250) — matrix operand order repaired and checked by reading, not compiled, not run on hardware
//
#include <hip/hip_runtime.h>
#include <hip/hip_bf16.h>

// ---------------------------------------------------------------------------
// CDNA5 / gfx1250 fused CapInfoNCE pipeline.
// Precision strategy: all GEMMs run on v_wmma_f32_16x16x32_bf16 (f32 accum),
// softmax / reductions in f32. Problem is HBM-bound (~560 MB traffic), so we
// fuse logits->softmax->PV->row-dot per (w,b) tile to write att exactly once.
// Tile staging into LDS uses the Tensor Data Mover (tensor_load_to_lds) with
// hand-built D# descriptors; TDM OOB-zero semantics provide row padding free.
// Fragment loads are expressed as 128-bit LDS reads (ds_load_b128).
// ---------------------------------------------------------------------------

typedef __attribute__((ext_vector_type(16))) __bf16 v16bf;
typedef __attribute__((ext_vector_type(8)))  float  v8f;
typedef __attribute__((ext_vector_type(4)))  unsigned int v4u;
typedef __attribute__((ext_vector_type(8)))  int  v8i_t;
typedef __attribute__((ext_vector_type(4)))  int  v4i_t;

union FragB16 { v16bf v; v4u q[2]; };
union AccF    { v8f   v; float f[8]; };

__device__ inline unsigned short f32_to_bf16(float x) {
  unsigned int b = __float_as_uint(x);
  unsigned int r = b + 0x7FFFu + ((b >> 16) & 1u);   // round-to-nearest-even
  return (unsigned short)(r >> 16);
}

// A-fragment (16x32 bf16, MxK) from LDS, row-major [m][k], pitch in elements.
// ISA layout: lanes 0-15 m=lane (half=0), lanes 16-31 m=lane-16 (half=1);
// v0..3 hold K = 2v+8*half pairs, v4..7 hold K = 16+2(v-4)+8*half pairs.
// Both 4-dword groups are contiguous & 16B aligned -> two ds_load_b128.
__device__ inline v16bf load_frag_a(const unsigned short* base, int m, int pitch,
                                    int kbase, int half) {
  FragB16 f;
  const v4u* p = (const v4u*)(base + m * pitch + kbase);
  f.q[0] = p[half];        // dwords kbase/2 + 4*half   .. +3  (K = kbase+8h..+7)
  f.q[1] = p[2 + half];    // dwords kbase/2 + 8 + 4*half .. +3
  return f.v;
}

// B-fragment (32x16 bf16, KxN) from LDS stored K-contiguous per column:
// storage [n][k]. Lanes 0-15: n=lane, K=0..7,16..23; lanes 16-31: K=8..15,24..31.
// two 4-dword groups (dwords 4*khalf.., 8+4*khalf..) -> two ds_load_b128.
__device__ inline v16bf load_frag_b(const unsigned short* base, int n, int pitch,
                                    int kbase, int khalf) {
  FragB16 f;
  const v4u* p = (const v4u*)(base + n * pitch + kbase);
  f.q[0] = p[khalf];
  f.q[1] = p[2 + khalf];
  return f.v;
}

__device__ inline v8f wmma_bf16(v16bf a, v16bf b, v8f c) {
  return __builtin_amdgcn_wmma_f32_16x16x32_bf16(false, a, false, b,
                                                 (short)0, c, false, false);
}

// ---------------------------------------------------------------------------
// TDM: 2D tile load Global -> LDS. D# per CDNA5 ISA.
// dataSizeLog: 0=1B,1=2B,2=4B. dims/strides/tiles in elements.
// OOB rows/cols (tile larger than tensor) read as zero -> free padding.
// Issue from ONE wave only (wave-uniform branch); TDM ignores EXEC.
// 6-arg builtin on this toolchain: (v4u, v8i, v4i, v4i, v8i, cpol);
// groups 2/3 (5D/gather descriptors) are unused for 2D tiles -> zero-filled.
// ---------------------------------------------------------------------------
__device__ inline void tdm_load_2d(const void* gsrc, unsigned ldsOff,
                                   unsigned dataSizeLog,
                                   unsigned dim0, unsigned dim1, unsigned stride0,
                                   unsigned tile0, unsigned tile1) {
  unsigned long long ga = (unsigned long long)gsrc;
  v4u g0;
  g0[0] = 1u;                                            // count=1, user mode
  g0[1] = ldsOff;                                        // lds_addr [63:32]
  g0[2] = (unsigned)(ga & 0xFFFFFFFFu);                  // global_addr lo
  g0[3] = (unsigned)((ga >> 32) & 0x01FFFFFFu) | (2u << 30); // addr hi | type=2
  v8i_t g1;
  g1[0] = (int)(dataSizeLog << 16);                      // wg_mask=0, data_size
  g1[1] = (int)(dim0 << 16);                             // tensor_dim0[15:0]
  g1[2] = (int)((dim0 >> 16) | (dim1 << 16));            // dim0 hi | dim1 lo
  g1[3] = (int)((dim1 >> 16) | (tile0 << 16));           // dim1 hi | tile_dim0
  g1[4] = (int)tile1;                                    // tile_dim1, tile_dim2=0
  g1[5] = (int)stride0;                                  // tensor_dim0_stride lo
  g1[6] = 0;                                             // stride hi | dim1_stride
  g1[7] = 0;
  v4i_t z4 = {0, 0, 0, 0};
  v8i_t z8 = {0, 0, 0, 0, 0, 0, 0, 0};
  __builtin_amdgcn_tensor_load_to_lds(g0, g1, z4, z4, z8, 0);
}

__device__ inline unsigned lds_offset(const void* p) {
  // Generic pointers into LDS carry the offset in the low 32 bits
  // (aperture mapping truncates to addr[31:0], ISA 10.2).
  return (unsigned)(unsigned long long)p;
}

// ---------------------------------------------------------------------------
// Projection GEMM: C[M x 256] = A[M x K] * W[K x 256] + bias.
// 64x64 tile per block, 8 waves, K-step 32. OUTF32=0 -> bf16 out, 1 -> f32.
// ---------------------------------------------------------------------------
template <int OUTF32>
__global__ __launch_bounds__(256) void gemm_n256(
    const float* __restrict__ A, const float* __restrict__ W,
    const float* __restrict__ bias, void* __restrict__ Cout,
    int M, int K) {
  __shared__ unsigned short sA[64 * 32];   // [m][k] bf16
  __shared__ unsigned short sW[64 * 32];   // [n][k] bf16 (K-transposed)

  const int tid   = threadIdx.x;
  const int mBase = blockIdx.x * 64;
  const int nBase = blockIdx.y * 64;
  const int wv    = tid >> 5;
  const int ln    = tid & 31;
  const int half  = ln >> 4;
  const int nl    = ln & 15;
  const int m_tile = wv & 3;          // 4 m-tiles
  const int n0     = (wv >> 2) * 2;   // 2 n-tiles per wave

  AccF acc0, acc1;
#pragma unroll
  for (int r = 0; r < 8; ++r) { acc0.f[r] = 0.f; acc1.f[r] = 0.f; }

  for (int k0 = 0; k0 < K; k0 += 32) {
#pragma unroll
    for (int j = 0; j < 8; ++j) {            // A tile 64x32
      int i = tid + 256 * j;
      int r = i >> 5, c = i & 31;
      sA[r * 32 + c] = f32_to_bf16(A[(size_t)(mBase + r) * K + k0 + c]);
    }
#pragma unroll
    for (int j = 0; j < 8; ++j) {            // W tile 32x64 -> [n][k]
      int i = tid + 256 * j;
      int kk = i >> 6, c = i & 63;
      sW[c * 32 + kk] = f32_to_bf16(W[(size_t)(k0 + kk) * 256 + nBase + c]);
    }
    if (k0 + 32 < K) {                        // prefetch next K-slice
#pragma unroll
      for (int j = 0; j < 8; ++j) {
        int i = tid + 256 * j;
        __builtin_prefetch(&A[(size_t)(mBase + (i >> 5)) * K + k0 + 32 + (i & 31)], 0, 1);
      }
      __builtin_prefetch(&W[(size_t)(k0 + 32 + (tid >> 6)) * 256 + nBase + (tid & 63)], 0, 1);
    }
    __syncthreads();

    v16bf a  = load_frag_a(sA, 16 * m_tile + nl, 32, 0, half);
    v16bf b0 = load_frag_b(sW, 16 * n0 + nl, 32, 0, half);
    v16bf b1 = load_frag_b(sW, 16 * (n0 + 1) + nl, 32, 0, half);
    acc0.v = wmma_bf16(a, b0, acc0.v);
    acc1.v = wmma_bf16(a, b1, acc1.v);
    __syncthreads();
  }

#pragma unroll
  for (int j = 0; j < 2; ++j) {
    AccF& acc = j ? acc1 : acc0;
    int ncol = nBase + 16 * (n0 + j) + nl;
    float bv = bias[ncol];
#pragma unroll
    for (int r = 0; r < 8; ++r) {
      int m = mBase + 16 * m_tile + r + 8 * half;
      float v = acc.f[r] + bv;
      if (OUTF32) ((float*)Cout)[(size_t)m * 256 + ncol] = v;
      else ((unsigned short*)Cout)[(size_t)m * 256 + ncol] = f32_to_bf16(v);
    }
  }
}

// ---------------------------------------------------------------------------
// Fused attention: one block per (w,b) in 96x96.
//   S(32x112)  = Kw[w](32x256) x Ku[b]^T   (K=256, 8 wmma k-steps)
//   att        = softmax(S/16) over To=100 -> written once to d_out
//   O(32x256)  = P(32x128 bf16) x V_o[b]   (K=128 zero-padded, 4 k-steps)
//   logits     = rowdot(O, V_w[w])
// Dynamic LDS layout (124 KB, 2 blocks/WGP within 320 KB):
//   [0, 73728)   : phase1: A1 Kw 32x256 bf16 (16KB) + B1 Ku 112x256 bf16 (56KB)
//                  phase4: VoT 256x128 bf16 (64KB);  phase6: O 32x256 f32 (32KB)
//   [73728, ..)  : S 32x112 f32 (14KB)
//   [88064, ..)  : P 32x128 bf16 (8KB)
//   [96256, ..)  : Vw 30x256 f32 (30KB)
// ---------------------------------------------------------------------------
#define SMEM_BYTES 126976

__global__ __launch_bounds__(256) void fused_att(
    const unsigned short* __restrict__ KwG, const unsigned short* __restrict__ KuG,
    const unsigned short* __restrict__ VoG, const float* __restrict__ VwG,
    float* __restrict__ attOut, float* __restrict__ avoOut,
    float* __restrict__ logitsOut) {
  extern __shared__ char smem[];
  unsigned short* A1  = (unsigned short*)(smem);           // 32x256 bf16
  unsigned short* B1  = (unsigned short*)(smem + 16384);   // 112x256 bf16
  unsigned short* VoT = (unsigned short*)(smem);           // 256x128 bf16
  float*          O   = (float*)(smem);                    // 32x256 f32
  float*          S   = (float*)(smem + 73728);            // 32x112 f32
  unsigned short* P   = (unsigned short*)(smem + 88064);   // 32x128 bf16
  float*          Vw  = (float*)(smem + 96256);            // 30x256 f32

  const int blk = blockIdx.x;
  const int w = blk / 96, b = blk % 96;
  const int tid = threadIdx.x;
  const int wv = tid >> 5, ln = tid & 31, half = ln >> 4, nl = ln & 15;
  const int g = tid >> 3, l8 = tid & 7;    // 8-lane row groups for reductions

  // ---- phase 1: TDM-stage Kw[w] (30x256 -> 32x256, OOB rows = 0),
  //      Ku[b] (100x256 -> 112x256, OOB rows = 0), Vw[w] (30x256 f32).
  //      Wave 0 issues the DMAs; everyone zeroes P meanwhile. ----
  if (wv == 0) {
    tdm_load_2d(KwG + (size_t)w * 30 * 256, lds_offset(A1),
                /*2B*/1, 256, 30, 256, 256, 32);
    tdm_load_2d(KuG + (size_t)b * 100 * 256, lds_offset(B1),
                /*2B*/1, 256, 100, 256, 256, 112);
    tdm_load_2d(VwG + (size_t)w * 30 * 256, lds_offset(Vw),
                /*4B*/2, 256, 30, 256, 256, 30);
  }
  {
    unsigned int* Pu = (unsigned int*)P;
    for (int i = tid; i < 32 * 64; i += 256) Pu[i] = 0u;
  }
  if (wv == 0) __builtin_amdgcn_s_wait_tensorcnt(0);
  __syncthreads();

  // ---- phase 2: S = Kw x Ku^T, scaled by 1/sqrt(256). 2x7 tiles, 8 waves ----
  for (int t = wv; t < 14; t += 8) {
    int mt = t / 7, nt = t % 7;
    AccF acc;
#pragma unroll
    for (int r = 0; r < 8; ++r) acc.f[r] = 0.f;
#pragma unroll
    for (int ks = 0; ks < 8; ++ks) {
      v16bf a  = load_frag_a(A1, 16 * mt + nl, 256, 32 * ks, half);
      v16bf bb = load_frag_b(B1, 16 * nt + nl, 256, 32 * ks, half);
      acc.v = wmma_bf16(a, bb, acc.v);
    }
#pragma unroll
    for (int r = 0; r < 8; ++r) {
      int mm = 16 * mt + r + 8 * half;
      S[mm * 112 + 16 * nt + nl] = acc.f[r] * 0.0625f;
    }
  }
  __syncthreads();

  // ---- phase 3: row softmax over To=100; write att; pack P to bf16 ----
  if (g < 30) {
    float mx = -3.4e38f;
    for (int c = l8; c < 100; c += 8) mx = fmaxf(mx, S[g * 112 + c]);
#pragma unroll
    for (int s = 1; s < 8; s <<= 1) mx = fmaxf(mx, __shfl_xor(mx, s, 8));
    float sum = 0.f;
    for (int c = l8; c < 100; c += 8) {
      float e = __expf(S[g * 112 + c] - mx);
      S[g * 112 + c] = e;
      sum += e;
    }
#pragma unroll
    for (int s = 1; s < 8; s <<= 1) sum += __shfl_xor(sum, s, 8);
    float inv = 1.f / sum;
    size_t attBase = ((size_t)blk * 30 + g) * 100;
    for (int c = l8; c < 100; c += 8) {
      float a = S[g * 112 + c] * inv;
      attOut[attBase + c] = a;
      P[g * 128 + c] = f32_to_bf16(a);
    }
  }

  // ---- phase 4: stage V_o[b] transposed (K-contiguous per column) ----
  {
    const unsigned short* src = VoG + (size_t)b * 100 * 256;
    for (int i = tid; i < 100 * 256; i += 256) {
      int k = i >> 8, n = i & 255;
      VoT[n * 128 + k] = src[i];
    }
    for (int i = tid; i < 28 * 256; i += 256) {   // zero K padding 100..127
      int k = 100 + (i >> 8), n = i & 255;
      VoT[n * 128 + k] = 0;
    }
  }
  __syncthreads();

  // ---- phase 5: O = P x V_o. 2x16 tiles, 4 per wave, 4 k-steps ----
  AccF oa[4];
#pragma unroll
  for (int q = 0; q < 4; ++q) {
    int t = wv + 8 * q;
    int mt = t >> 4, nt = t & 15;
#pragma unroll
    for (int r = 0; r < 8; ++r) oa[q].f[r] = 0.f;
#pragma unroll
    for (int ks = 0; ks < 4; ++ks) {
      v16bf a  = load_frag_a(P, 16 * mt + nl, 128, 32 * ks, half);
      v16bf bb = load_frag_b(VoT, 16 * nt + nl, 128, 32 * ks, half);
      oa[q].v = wmma_bf16(a, bb, oa[q].v);
    }
  }
  __syncthreads();                      // everyone done reading VoT region
#pragma unroll
  for (int q = 0; q < 4; ++q) {
    int t = wv + 8 * q;
    int mt = t >> 4, nt = t & 15;
#pragma unroll
    for (int r = 0; r < 8; ++r)
      O[(16 * mt + r + 8 * half) * 256 + 16 * nt + nl] = oa[q].f[r];
  }
  __syncthreads();

  // ---- phase 6: coalesced att_V_o store + logits row-dots with Vw ----
  {
    size_t avoBase = (size_t)blk * 30 * 256;
    for (int i = tid; i < 30 * 256; i += 256) avoOut[avoBase + i] = O[i];
  }
  if (g < 30) {
    float p = 0.f;
    for (int d = l8; d < 256; d += 8) p += O[g * 256 + d] * Vw[g * 256 + d];
#pragma unroll
    for (int s = 1; s < 8; s <<= 1) p += __shfl_xor(p, s, 8);
    if (l8 == 0) logitsOut[(size_t)blk * 30 + g] = p;
  }
}

// ---------------------------------------------------------------------------
// Loss: only the b==w diagonal of lsm contributes. Single block, tree reduce.
// ---------------------------------------------------------------------------
__global__ __launch_bounds__(256) void loss_kernel(
    const float* __restrict__ logits, const float* __restrict__ mask,
    float* __restrict__ out) {
  __shared__ float red[256];
  const int tid = threadIdx.x;
  float acc = 0.f;
  for (int i = tid; i < 96 * 30; i += 256) {
    int w = i / 30, t = i % 30;
    float mx = -3.4e38f;
    for (int b = 0; b < 96; ++b)
      mx = fmaxf(mx, logits[((size_t)w * 96 + b) * 30 + t]);
    float s = 0.f;
    for (int b = 0; b < 96; ++b)
      s += __expf(logits[((size_t)w * 96 + b) * 30 + t] - mx);
    float lse = mx + __logf(s);
    float diag = logits[((size_t)w * 96 + w) * 30 + t] - lse;
    float nnm = 0.f;
    for (int tt = 0; tt < 30; ++tt) nnm += 1.f - mask[w * 30 + tt];
    acc += (1.f - mask[w * 30 + t]) * diag / (nnm + 1e-6f);
  }
  red[tid] = acc;
  __syncthreads();
  for (int s = 128; s > 0; s >>= 1) {
    if (tid < s) red[tid] += red[tid + s];
    __syncthreads();
  }
  if (tid == 0) out[0] = -red[0] / 96.f;
}

// ---------------------------------------------------------------------------
extern "C" void kernel_launch(void* const* d_in, const int* in_sizes, int n_in,
                              void* d_out, int out_size, void* d_ws, size_t ws_size,
                              hipStream_t stream) {
  const float* o    = (const float*)d_in[0];
  const float* u    = (const float*)d_in[1];
  const float* w    = (const float*)d_in[2];
  const float* mask = (const float*)d_in[3];
  const float* Wku  = (const float*)d_in[4];
  const float* bku  = (const float*)d_in[5];
  const float* Wkw  = (const float*)d_in[6];
  const float* bkw  = (const float*)d_in[7];
  const float* Wfo  = (const float*)d_in[8];
  const float* bfo  = (const float*)d_in[9];
  const float* Wfw  = (const float*)d_in[10];
  const float* bfw  = (const float*)d_in[11];

  char* ws = (char*)d_ws;
  size_t off = 0;
  unsigned short* Ku = (unsigned short*)(ws + off); off += (size_t)9600 * 256 * 2;
  unsigned short* Kw = (unsigned short*)(ws + off); off += (size_t)2880 * 256 * 2;
  unsigned short* Vo = (unsigned short*)(ws + off); off += (size_t)9600 * 256 * 2;
  float*          Vw = (float*)(ws + off);          off += (size_t)2880 * 256 * 4;
  float*      logits = (float*)(ws + off);          // 96*96*30 f32

  float* outF = (float*)d_out;
  float* loss = outF;
  float* att  = outF + 1;
  float* avo  = outF + 1 + (size_t)96 * 96 * 30 * 100;

  dim3 blk(256);
  gemm_n256<0><<<dim3(9600 / 64, 4), blk, 0, stream>>>(u, Wku, bku, Ku, 9600, 2048);
  gemm_n256<0><<<dim3(2880 / 64, 4), blk, 0, stream>>>(w, Wkw, bkw, Kw, 2880, 768);
  gemm_n256<0><<<dim3(9600 / 64, 4), blk, 0, stream>>>(o, Wfo, bfo, Vo, 9600, 1024);
  gemm_n256<1><<<dim3(2880 / 64, 4), blk, 0, stream>>>(w, Wfw, bfw, Vw, 2880, 768);
  fused_att<<<dim3(96 * 96), blk, SMEM_BYTES, stream>>>(Kw, Ku, Vo, Vw, att, avo, logits);
  loss_kernel<<<dim3(1), blk, 0, stream>>>(logits, mask, loss);
}
